// RelAttnBlock_82523501625504
// MI455X (gfx1250) — matrix operand order repaired and checked by reading, not compiled, not run on hardware
//
#include <hip/hip_runtime.h>
#include <hip/hip_bf16.h>

// ---------------- types ----------------
typedef __bf16 bf16_t;
typedef __attribute__((ext_vector_type(16))) __bf16 v16bf;
typedef __attribute__((ext_vector_type(8)))  __bf16 v8bf;
typedef __attribute__((ext_vector_type(8)))  float  v8f;

#define EMB 512
#define NB  8
#define LSEQ 1024
#define NH  8
#define HD  64
#define BSREL 1024
#define NTOK (NB * LSEQ)   // 8192

// float -> bf16, round-to-nearest-even
__device__ __forceinline__ bf16_t f2bf(float f) {
    unsigned u = __float_as_uint(f);
    unsigned r = (u + 0x7FFFu + ((u >> 16) & 1u)) >> 16;
    return __builtin_bit_cast(bf16_t, (unsigned short)r);
}

__device__ __forceinline__ v8f wmma_bf16(v16bf a, v16bf b, v8f c) {
    return __builtin_amdgcn_wmma_f32_16x16x32_bf16(
        /*neg_a=*/false, a, /*neg_b=*/false, b,
        /*c_mod=*/(short)0, c, /*reuse_a=*/false, /*reuse_b=*/false);
}

// A fragment: 16x32 bf16 tile of row-major A (lda elems), at (row0, k0).
// lane m = lane&15 ; k-half = lane>>4 ; elems {kh*16+0..15}
// Works for global or LDS-backed (generic) pointers.
__device__ __forceinline__ v16bf load_a_frag(const bf16_t* A, int lda, int row0,
                                             int k0, int lane) {
    int m  = lane & 15;
    int kh = lane >> 4;
    const bf16_t* p = A + (size_t)(row0 + m) * lda + k0 + kh * 16;
    v8bf lo = *(const v8bf*)p;          // k = k0 + kh*16 + 0..7
    v8bf hi = *(const v8bf*)(p + 8);    // k = k0 + kh*16 + 8..15
    v16bf r;
#pragma unroll
    for (int i = 0; i < 8; ++i) { r[i] = lo[i]; r[i + 8] = hi[i]; }
    return r;
}

// B fragment from the N-major transpose: M[n][k] (ldm elems/row).
// B[kk][n] = M[n0+lane%16][k0 + (lane/16)*16 + kk']; contiguous 32-B vector load.
__device__ __forceinline__ v16bf load_b_frag_nmajor(const bf16_t* Mm, int ldm,
                                                    int n0, int k0, int lane) {
    int n  = n0 + (lane & 15);
    int kb = k0 + ((lane >> 4) << 4);
    return *(const v16bf*)(Mm + (size_t)n * ldm + kb);
}

// ---------------- small kernels ----------------
__global__ void cvt_bf16_kernel(const float* __restrict__ src,
                                bf16_t* __restrict__ dst, int n) {
    int i = blockIdx.x * blockDim.x + threadIdx.x;
    if (i < n) dst[i] = f2bf(src[i]);
}

// W (512x512 row-major [k][n]) -> bf16 transposed [n][k]
__global__ void cvt_transpose_bf16_kernel(const float* __restrict__ src,
                                          bf16_t* __restrict__ dst) {
    int i = blockIdx.x * blockDim.x + threadIdx.x;   // 512*512 threads
    int k = i >> 9, n = i & (EMB - 1);
    dst[(size_t)n * EMB + k] = f2bf(src[i]);
}

__global__ void layernorm_kernel(const float* __restrict__ x,
                                 const float* __restrict__ g,
                                 const float* __restrict__ bta,
                                 bf16_t* __restrict__ hout) {
    int t = blockIdx.x;                     // token 0..8191
    int tid = threadIdx.x;                  // 256 threads, 2 elems each
    const float* row = x + (size_t)t * EMB;
    float v0 = row[tid], v1 = row[tid + 256];
    __shared__ float red[256];
    red[tid] = v0 + v1;
    __syncthreads();
#pragma unroll
    for (int off = 128; off > 0; off >>= 1) {
        if (tid < off) red[tid] += red[tid + off];
        __syncthreads();
    }
    float mu = red[0] * (1.0f / EMB);
    __syncthreads();
    float d0 = v0 - mu, d1 = v1 - mu;
    red[tid] = d0 * d0 + d1 * d1;
    __syncthreads();
#pragma unroll
    for (int off = 128; off > 0; off >>= 1) {
        if (tid < off) red[tid] += red[tid + off];
        __syncthreads();
    }
    float rs = rsqrtf(red[0] * (1.0f / EMB) + 1e-5f);
    hout[(size_t)t * EMB + tid]       = f2bf(d0 * rs * g[tid] + bta[tid]);
    hout[(size_t)t * EMB + tid + 256] = f2bf(d1 * rs * g[tid + 256] + bta[tid + 256]);
}

// ---- WMMA GEMM: C(8192x512) = A(8192x512) @ Bt^T + bias ; Bt is N-major [n][k]
// MODE 0: bf16 to (B,H,L,HD) split-head. MODE 1: f32 row-major.
// MODE 2: bf16 to per-head transposed V: vT[((b*NH+h)*HD + d)*LSEQ + l].
// Block: 256 threads = 8 waves; block tile 32x256; per-wave tile 16x64.
template <int MODE>
__global__ void gemm_kernel(const bf16_t* __restrict__ A,
                            const bf16_t* __restrict__ Bt,
                            const float* __restrict__ bias,
                            void* __restrict__ dstv) {
    const int tid = threadIdx.x;
    const int wave = tid >> 5, lane = tid & 31;
    const int m0 = blockIdx.x * 32 + (wave >> 2) * 16;
    const int n0 = blockIdx.y * 256 + (wave & 3) * 64;
    v8f acc[4] = {v8f{}, v8f{}, v8f{}, v8f{}};
#pragma unroll 2
    for (int k0 = 0; k0 < EMB; k0 += 32) {
        if (k0 + 128 < EMB)   // gfx1250 global_prefetch_b8 of next A chunk
            __builtin_prefetch(A + (size_t)(m0 + (lane & 15)) * EMB + k0 + 128, 0, 1);
        v16bf a = load_a_frag(A, EMB, m0, k0, lane);
#pragma unroll
        for (int t = 0; t < 4; ++t) {
            v16bf b = load_b_frag_nmajor(Bt, EMB, n0 + t * 16, k0, lane);
            acc[t] = wmma_bf16(a, b, acc[t]);
        }
    }
    const int nlane = lane & 15, rowsel = lane >> 4;
#pragma unroll
    for (int t = 0; t < 4; ++t) {
        int n = n0 + t * 16 + nlane;
        float bv = bias[n];
#pragma unroll
        for (int r = 0; r < 8; ++r) {
            int m = m0 + r + 8 * rowsel;
            float val = acc[t][r] + bv;
            int bb = m >> 10, l = m & (LSEQ - 1);
            int hh = n >> 6, d = n & (HD - 1);
            if (MODE == 0) {        // (B,H,L,HD) bf16
                ((bf16_t*)dstv)[((size_t)(bb * NH + hh) * LSEQ + l) * HD + d] = f2bf(val);
            } else if (MODE == 2) { // (B,H,HD,L) bf16 (V transposed per head)
                ((bf16_t*)dstv)[((size_t)(bb * NH + hh) * HD + d) * LSEQ + l] = f2bf(val);
            } else {                // (B,L,EMB) f32
                ((float*)dstv)[(size_t)m * EMB + n] = val;
            }
        }
    }
}

// ---------------- fused attention kernel -------------------------------------
// grid = B*H*(L/16) = 4096 blocks, 128 threads (4 waves), ~129 KB dynamic LDS.
__global__ void attn_kernel(const bf16_t* __restrict__ qb,
                            const bf16_t* __restrict__ kb,
                            const bf16_t* __restrict__ vTb,
                            const bf16_t* __restrict__ erb,
                            bf16_t* __restrict__ attn_out) {
    extern __shared__ float smem[];
    float* qer    = smem;                  // 16*1024 f32 (reused as bf16 exp later)
    float* sc     = qer + 16 * 1024;       // 16*1024 f32
    float* pr     = sc + 16 * 1024;        // 16*8 partials
    float* rowmax = pr + 16 * 8;           // 16
    float* rowinv = rowmax + 16;           // 16
    bf16_t* wexp  = (bf16_t*)qer;          // bf16 [16][1024] exp weights (32 KB)

    const int blk = blockIdx.x;
    const int lt = blk & 63, hh = (blk >> 6) & 7, bb = blk >> 9;
    const int l0 = lt * 16;
    const int tid = threadIdx.x, wave = tid >> 5, lane = tid & 31;
    const int nlane = lane & 15, rowsel = lane >> 4;

    const bf16_t* qbase  = qb  + (size_t)(bb * NH + hh) * LSEQ * HD;
    const bf16_t* kbase  = kb  + (size_t)(bb * NH + hh) * LSEQ * HD;
    const bf16_t* vTbase = vTb + (size_t)(bb * NH + hh) * HD * LSEQ;
    const bf16_t* ebase  = erb + (size_t)hh * BSREL * HD;

    // q-tile A fragments (K = HD = 64 -> two 32-wide fragments), reused
    v16bf a0 = load_a_frag(qbase, HD, l0, 0, lane);
    v16bf a1 = load_a_frag(qbase, HD, l0, 32, lane);

    // Phase 1: QEr strip = q_tile(16x64) @ Er[h]^T(64x1024) -> LDS
#pragma unroll 2
    for (int j = 0; j < 16; ++j) {
        int nt = wave * 16 + j;
        v8f acc = {};
        acc = wmma_bf16(a0, load_b_frag_nmajor(ebase, HD, nt * 16, 0, lane), acc);
        acc = wmma_bf16(a1, load_b_frag_nmajor(ebase, HD, nt * 16, 32, lane), acc);
#pragma unroll
        for (int r = 0; r < 8; ++r)
            qer[(r + 8 * rowsel) * 1024 + nt * 16 + nlane] = acc[r];
    }
    __syncthreads();

    // Phase 2: scores strip = scale * q k^T + shifted-gather(QEr) -> LDS
#pragma unroll 2
    for (int j = 0; j < 16; ++j) {
        int kt = wave * 16 + j;
        v8f acc = {};
        acc = wmma_bf16(a0, load_b_frag_nmajor(kbase, HD, kt * 16, 0, lane), acc);
        acc = wmma_bf16(a1, load_b_frag_nmajor(kbase, HD, kt * 16, 32, lane), acc);
        int kg = kt * 16 + nlane;
#pragma unroll
        for (int r = 0; r < 8; ++r) {
            int m = r + 8 * rowsel;
            int rel = (kg - (l0 + m) + BSREL) & (BSREL - 1);   // (k-l) mod BS
            sc[m * 1024 + kg] = acc[r] * 0.125f + qer[m * 1024 + rel];
        }
    }
    __syncthreads();

    // Phase 3: row softmax; write bf16 exp weights into wexp (qer is dead now)
    {
        int row = tid >> 3, sub = tid & 7;
        float* srow = sc + row * 1024;
        float mx = -3.0e38f;
        for (int i = sub * 128; i < sub * 128 + 128; ++i) mx = fmaxf(mx, srow[i]);
        pr[row * 8 + sub] = mx;
        __syncthreads();
        if (sub == 0) {
            float m2 = pr[row * 8];
#pragma unroll
            for (int i = 1; i < 8; ++i) m2 = fmaxf(m2, pr[row * 8 + i]);
            rowmax[row] = m2;
        }
        __syncthreads();
        float rm = rowmax[row];
        bf16_t* wrow = wexp + row * 1024;
        float sum = 0.f;
        for (int i = sub * 128; i < sub * 128 + 128; ++i) {
            float e = __expf(srow[i] - rm);
            wrow[i] = f2bf(e);
            sum += e;
        }
        pr[row * 8 + sub] = sum;
        __syncthreads();
        if (sub == 0) {
            float s2 = 0.f;
#pragma unroll
            for (int i = 0; i < 8; ++i) s2 += pr[row * 8 + i];
            rowinv[row] = 1.0f / s2;
        }
        __syncthreads();
    }

    // Phase 4: out = exp_strip(16x1024) @ v(1024x64); wave owns one 16-wide d-tile.
    // A fragments come straight from LDS bf16 (ds_load_b128 pairs); B fragments
    // are contiguous vector loads from per-head transposed V.
    {
        int n0 = wave * 16;
        v8f acc = {};
#pragma unroll 2
        for (int k0 = 0; k0 < LSEQ; k0 += 32) {
            v16bf a = load_a_frag((const bf16_t*)wexp, 1024, 0, k0, lane);
            v16bf b = load_b_frag_nmajor(vTbase, LSEQ, n0, k0, lane);
            acc = wmma_bf16(a, b, acc);
        }
#pragma unroll
        for (int r = 0; r < 8; ++r) {
            int m = r + 8 * rowsel;
            float val = acc[r] * rowinv[m];
            int tok = bb * LSEQ + l0 + m;
            attn_out[(size_t)tok * EMB + hh * HD + n0 + nlane] = f2bf(val);
        }
    }
}

// ---------------- launcher ----------------------------------------------------
extern "C" void kernel_launch(void* const* d_in, const int* in_sizes, int n_in,
                              void* d_out, int out_size, void* d_ws, size_t ws_size,
                              hipStream_t stream) {
    (void)in_sizes; (void)n_in; (void)out_size; (void)ws_size;
    const float* x    = (const float*)d_in[0];
    const float* ln_g = (const float*)d_in[1];
    const float* ln_b = (const float*)d_in[2];
    const float* Wq   = (const float*)d_in[3];
    const float* bq   = (const float*)d_in[4];
    const float* Wk   = (const float*)d_in[5];
    const float* bk   = (const float*)d_in[6];
    const float* Wv   = (const float*)d_in[7];
    const float* bv   = (const float*)d_in[8];
    const float* Wo   = (const float*)d_in[9];
    const float* bo   = (const float*)d_in[10];
    const float* Er   = (const float*)d_in[11];
    float* out = (float*)d_out;

    char* ws = (char*)d_ws;
    size_t off = 0;
    auto carve = [&](size_t bytes) -> void* {
        void* p = ws + off;
        off = (off + bytes + 255) & ~(size_t)255;
        return p;
    };
    bf16_t* h_bf   = (bf16_t*)carve((size_t)NTOK * EMB * 2);
    bf16_t* WqT_bf = (bf16_t*)carve((size_t)EMB * EMB * 2);
    bf16_t* WkT_bf = (bf16_t*)carve((size_t)EMB * EMB * 2);
    bf16_t* WvT_bf = (bf16_t*)carve((size_t)EMB * EMB * 2);
    bf16_t* WoT_bf = (bf16_t*)carve((size_t)EMB * EMB * 2);
    bf16_t* Er_bf  = (bf16_t*)carve((size_t)NH * BSREL * HD * 2);
    bf16_t* q_bf   = (bf16_t*)carve((size_t)NTOK * EMB * 2);
    bf16_t* k_bf   = (bf16_t*)carve((size_t)NTOK * EMB * 2);
    bf16_t* vT_bf  = (bf16_t*)carve((size_t)NTOK * EMB * 2);
    bf16_t* at_bf  = (bf16_t*)carve((size_t)NTOK * EMB * 2);

    // 1) weight conversion: transpose to N-major bf16; Er stays r-major (k-contig)
    const int nE = NH * BSREL * HD;
    cvt_transpose_bf16_kernel<<<EMB * EMB / 256, 256, 0, stream>>>(Wq, WqT_bf);
    cvt_transpose_bf16_kernel<<<EMB * EMB / 256, 256, 0, stream>>>(Wk, WkT_bf);
    cvt_transpose_bf16_kernel<<<EMB * EMB / 256, 256, 0, stream>>>(Wv, WvT_bf);
    cvt_transpose_bf16_kernel<<<EMB * EMB / 256, 256, 0, stream>>>(Wo, WoT_bf);
    cvt_bf16_kernel<<<(nE + 255) / 256, 256, 0, stream>>>(Er, Er_bf, nE);

    // 2) LayerNorm -> bf16 h
    layernorm_kernel<<<NTOK, 256, 0, stream>>>(x, ln_g, ln_b, h_bf);

    // 3) Q/K/V projections (WMMA GEMM); V is written per-head transposed
    dim3 gg(NTOK / 32, EMB / 256);
    gemm_kernel<0><<<gg, 256, 0, stream>>>(h_bf, WqT_bf, bq, (void*)q_bf);
    gemm_kernel<0><<<gg, 256, 0, stream>>>(h_bf, WkT_bf, bk, (void*)k_bf);
    gemm_kernel<2><<<gg, 256, 0, stream>>>(h_bf, WvT_bf, bv, (void*)vT_bf);

    // 4) fused relative attention
    size_t shmem = (size_t)(16 * 1024 * 2 + 16 * 8 + 16 + 16) * sizeof(float);
    attn_kernel<<<NB * NH * (LSEQ / 16), 128, shmem, stream>>>(q_bf, k_bf, vT_bf,
                                                               Er_bf, at_bf);

    // 5) output projection -> f32 d_out
    gemm_kernel<1><<<gg, 256, 0, stream>>>(at_bf, WoT_bf, bo, (void*)out);
}
